// RnnReceiverFixedLengthGS_10325101379877
// MI455X (gfx1250) — matrix operand order repaired and checked by reading, not compiled, not run on hardware
//
#include <hip/hip_runtime.h>
#include <math.h>

// ---------------------------------------------------------------------------
// MI455X (gfx1250) GRU receiver.
//  * All GEMMs via v_wmma_f32_16x16x32_bf16 (f32 accum).
//  * ~347 GFLOP vs ~160MB unique traffic -> matrix-engine bound => bf16 WMMA.
//  * bf16-A GEMMs stage their A tile with the Tensor Data Mover
//    (tensor_load_to_lds + s_wait_tensorcnt), using D# LDS padding to
//    reproduce the bank-conflict-padded LDS layout.
// ---------------------------------------------------------------------------

typedef __bf16 bf16;
typedef __attribute__((ext_vector_type(16))) __bf16 v16bf;
typedef __attribute__((ext_vector_type(8)))  float  v8f;

typedef __attribute__((ext_vector_type(4))) unsigned int tdm_u4;
typedef __attribute__((ext_vector_type(8))) int          tdm_i8;
typedef __attribute__((ext_vector_type(4))) int          tdm_i4;

#define BM 128
#define BN 128
#define BK 64
#define LDA_S 72    // BK + 8 pad (bank spread for b128 A-frag reads)
#define LDB_S 136   // BN + 8 pad (bank spread for b128 B-frag reads)

union FragU { v16bf v; uint4 q[2]; };

// TDM: DMA a BMxBK bf16 tile (row stride lda elements) from global into LDS
// at lds_off, inserting an 8-bf16 (4-DWORD) pad after every 64-bf16 row so
// the LDS image matches As[BM][LDA_S].  D# layout per CDNA5 ISA.
// This toolchain exposes the 6-arg builtin (clang-23 / therock-10.0 form).
__device__ __forceinline__ void tdm_load_a_tile(const bf16* gptr, unsigned lds_off, int lda) {
    unsigned long long ga = (unsigned long long)(size_t)gptr;
    tdm_u4 g0;
    g0.x = 0x1u;                                        // count=1 (valid), user mode
    g0.y = lds_off;                                     // lds_addr
    g0.z = (unsigned)ga;                                // global_addr[31:0]
    g0.w = (unsigned)((ga >> 32) & 0x1FFFFFFu) | (2u << 30);  // addr[56:32] | type=2
    tdm_i8 g1;
    g1[0] = (1 << 16)        // data_size: 2 bytes
          | (1 << 20)        // pad_enable
          | (4 << 22)        // pad_interval: 32 DWORDs (= one 64-bf16 row)
          | (3 << 25);       // pad_amount: 4 DWORDs (= 8 bf16)
    g1[1] = (BK << 16);      // tensor_dim0[15:0] = 64  (tensor==tile, never OOB)
    g1[2] = (BM << 16);      // tensor_dim0[31:16]=0 | tensor_dim1[15:0] = 128
    g1[3] = (BK << 16);      // tensor_dim1[31:16]=0 | tile_dim0 = 64
    g1[4] = BM;              // tile_dim1 = 128 | tile_dim2 = 0
    g1[5] = lda;             // tensor_dim0_stride[31:0]
    g1[6] = 0;               // stride[47:32]=0 | tensor_dim1_stride lo (unused, 2D)
    g1[7] = 0;
    tdm_i4 z4 = {0, 0, 0, 0};                           // 2D: groups 2/3 zero
    tdm_i8 z8 = {0, 0, 0, 0, 0, 0, 0, 0};
    __builtin_amdgcn_tensor_load_to_lds(g0, g1, z4, z4, z8, 0);
}

// C[M,N] (f32 or bf16) = A[M,K] @ W[N,K]^T + bias[N]
// A: f32 or bf16 row-major (lda), W: bf16 row-major [N,K], grid covers M,N
// exactly (all dims here are multiples of 128; K multiple of 64).
template <typename AT, bool OUT_BF16>
__global__ __launch_bounds__(256)
void wmma_gemm(const AT* __restrict__ A, int lda,
               const bf16* __restrict__ Wb,
               const float* __restrict__ bias,
               float* __restrict__ Cf, bf16* __restrict__ Cb,
               int ldc, int K)
{
    __shared__ __align__(16) bf16 As[BM][LDA_S];
    __shared__ __align__(16) bf16 Bs[BK][LDB_S];

    const int tid  = threadIdx.x;
    const int lane = tid & 31;
    const int wave = tid >> 5;
    const int wm   = (wave & 3) * 32;   // wave tile: 32 rows
    const int wn   = (wave >> 2) * 64;  //            64 cols
    const int mBlock = blockIdx.y * BM;
    const int nBlock = blockIdx.x * BN;

    constexpr bool A_IS_BF16 = __is_same(AT, bf16);
    const unsigned asOff = (unsigned)(size_t)(&As[0][0]);

    // A-fragment lane mapping (16x32 bf16, ISA 7.12.2):
    //   M = lane&15 ; lane<16: elems[0..7]=K0..7, [8..15]=K16..23
    //                 lane>=16: elems[0..7]=K8..15, [8..15]=K24..31
    const int mA   = lane & 15;
    const int khiA = (lane >> 4) * 8;

    v8f acc[2][4] = {};

    for (int k0 = 0; k0 < K; k0 += BK) {
        // ---- stage A tile ----
        if constexpr (A_IS_BF16) {
            if (wave == 0)   // one DMA per workgroup (TDM ignores EXEC; wave-uniform branch)
                tdm_load_a_tile((const bf16*)A + (size_t)mBlock * lda + k0, asOff, lda);
        } else {
            for (int i = tid; i < BM * BK; i += 256) {   // f32 -> bf16 convert path
                int r = i >> 6, c = i & (BK - 1);
                As[r][c] = (bf16)(float)A[(size_t)(mBlock + r) * lda + (k0 + c)];
            }
        }
        // ---- stage W tile transposed: Bs[k][n] = W[nBlock+n][k0+k] ----
        for (int i = tid; i < BN * BK; i += 256) {
            int n = i >> 6, k = i & (BK - 1);
            Bs[k][n] = Wb[(size_t)(nBlock + n) * K + (k0 + k)];
        }
        if (k0 + BK < K)     // global_prefetch_b8 of next W tile
            __builtin_prefetch(&Wb[(size_t)(nBlock + (tid & 127)) * K + (k0 + BK)], 0, 1);
        if constexpr (A_IS_BF16)
            __builtin_amdgcn_s_wait_tensorcnt(0);   // no-op for non-issuing waves
        __syncthreads();

        #pragma unroll
        for (int kk = 0; kk < BK; kk += 32) {
            FragU a0, a1;
            a0.q[0] = *(const uint4*)&As[wm +  0 + mA][kk +      khiA];
            a0.q[1] = *(const uint4*)&As[wm +  0 + mA][kk + 16 + khiA];
            a1.q[0] = *(const uint4*)&As[wm + 16 + mA][kk +      khiA];
            a1.q[1] = *(const uint4*)&As[wm + 16 + mA][kk + 16 + khiA];
            #pragma unroll
            for (int j = 0; j < 4; ++j) {
                // B-fragment (32x16 KxN): N = lane&15, elems follow the same
                // K order as A (table order) -> column gather from Bs.
                FragU b;
                #pragma unroll
                for (int s = 0; s < 16; ++s)
                    b.v[s] = Bs[kk + khiA + (s & 7) + ((s >> 3) << 4)][wn + j * 16 + mA];
                acc[0][j] = __builtin_amdgcn_wmma_f32_16x16x32_bf16(
                    false, a0.v, false, b.v, (short)0, acc[0][j], false, false);
                acc[1][j] = __builtin_amdgcn_wmma_f32_16x16x32_bf16(
                    false, a1.v, false, b.v, (short)0, acc[1][j], false, false);
            }
        }
        __syncthreads();
    }

    // D layout (16x16 f32): VGPR r -> M = (lane>>4)*8 + r, N = lane&15
    const int rowHalf = (lane >> 4) * 8;
    const int colLane = lane & 15;
    #pragma unroll
    for (int j = 0; j < 4; ++j) {
        int col = nBlock + wn + j * 16 + colLane;
        float bv = bias[col];
        #pragma unroll
        for (int i = 0; i < 2; ++i) {
            int rowBase = mBlock + wm + i * 16 + rowHalf;
            #pragma unroll
            for (int r = 0; r < 8; ++r) {
                float v = acc[i][j][r] + bv;
                size_t o = (size_t)(rowBase + r) * ldc + col;
                if constexpr (OUT_BF16) Cb[o] = (bf16)v;
                else                    Cf[o] = v;
            }
        }
    }
}

__global__ void cvt_f32_bf16(const float* __restrict__ s, bf16* __restrict__ d, int n) {
    int i = blockIdx.x * 256 + threadIdx.x;
    if (i < n) d[i] = (bf16)s[i];
}

__global__ void zero_h(float* __restrict__ h, bf16* __restrict__ hb, int n) {
    int i = blockIdx.x * 256 + threadIdx.x;
    if (i < n) { h[i] = 0.0f; hb[i] = (bf16)0.0f; }
}

// torch GRUCell gate order r,z,n: chunks [0:H),[H:2H),[2H:3H)
__global__ void gru_gate(const float* __restrict__ gi, const float* __restrict__ gh,
                         float* __restrict__ h, bf16* __restrict__ hb, int H) {
    int idx = blockIdx.x * 256 + threadIdx.x;   // over B*H
    int m = idx / H, n = idx - m * H;
    const float* gim = gi + (size_t)m * 3 * H;
    const float* ghm = gh + (size_t)m * 3 * H;
    float ir = gim[n], iz = gim[n + H], in_ = gim[n + 2 * H];
    float hr = ghm[n], hz = ghm[n + H], hn  = ghm[n + 2 * H];
    float r  = 1.0f / (1.0f + __expf(-(ir + hr)));
    float z  = 1.0f / (1.0f + __expf(-(iz + hz)));
    float nt = tanhf(in_ + r * hn);
    float hp = h[idx];
    float hnew = (1.0f - z) * nt + z * hp;
    h[idx]  = hnew;
    hb[idx] = (bf16)hnew;
}

extern "C" void kernel_launch(void* const* d_in, const int* in_sizes, int n_in,
                              void* d_out, int out_size, void* d_ws, size_t ws_size,
                              hipStream_t stream) {
    (void)in_sizes; (void)n_in; (void)out_size; (void)ws_size;
    constexpr int Bv = 1024, Tv = 32, Vv = 1024, Ev = 512, Hv = 1024, OUTv = 1024;

    const float* message    = (const float*)d_in[0];
    const float* emb_W      = (const float*)d_in[1];
    const float* emb_bias   = (const float*)d_in[2];
    const float* w_ih       = (const float*)d_in[3];
    const float* w_hh       = (const float*)d_in[4];
    const float* b_ih       = (const float*)d_in[5];
    const float* b_hh       = (const float*)d_in[6];
    const float* agent_W    = (const float*)d_in[7];
    const float* agent_bias = (const float*)d_in[8];

    char* ws = (char*)d_ws;
    size_t off = 0;
    auto abf  = [&](size_t n) { bf16*  p = (bf16*)(ws + off);  off += n * sizeof(bf16);  return p; };
    auto af32 = [&](size_t n) { float* p = (float*)(ws + off); off += n * sizeof(float); return p; };

    bf16* embW_b   = abf((size_t)Ev * Vv);        // 0.5M
    bf16* wih_b    = abf((size_t)3 * Hv * Ev);    // 1.5M
    bf16* whh_b    = abf((size_t)3 * Hv * Hv);    // 3M
    bf16* agentW_b = abf((size_t)OUTv * Hv);      // 1M
    bf16* emb16    = abf((size_t)Bv * Tv * Ev);   // 16M
    bf16* h16      = abf((size_t)Bv * Hv);        // 1M
    float* hf      = af32((size_t)Bv * Hv);
    float* gi      = af32((size_t)Bv * 3 * Hv);
    float* gh      = af32((size_t)Bv * 3 * Hv);   // total ~78 MB

    dim3 blk(256);

    cvt_f32_bf16<<<dim3((Ev * Vv) / 256),     blk, 0, stream>>>(emb_W,   embW_b,   Ev * Vv);
    cvt_f32_bf16<<<dim3((3 * Hv * Ev) / 256), blk, 0, stream>>>(w_ih,    wih_b,    3 * Hv * Ev);
    cvt_f32_bf16<<<dim3((3 * Hv * Hv) / 256), blk, 0, stream>>>(w_hh,    whh_b,    3 * Hv * Hv);
    cvt_f32_bf16<<<dim3((OUTv * Hv) / 256),   blk, 0, stream>>>(agent_W, agentW_b, OUTv * Hv);
    zero_h<<<dim3((Bv * Hv) / 256), blk, 0, stream>>>(hf, h16, Bv * Hv);

    // emb[B*T, E] = message @ emb_W^T + emb_b  (bf16 out, f32 A convert path)
    wmma_gemm<float, true><<<dim3(Ev / BN, (Bv * Tv) / BM), blk, 0, stream>>>(
        message, Vv, embW_b, emb_bias, nullptr, emb16, Ev, Vv);

    for (int t = 0; t < Tv; ++t) {
        // gi = emb[:, t, :] @ w_ih^T + b_ih   (rows strided by T*E; TDM A path)
        wmma_gemm<bf16, false><<<dim3(3 * Hv / BN, Bv / BM), blk, 0, stream>>>(
            emb16 + (size_t)t * Ev, Tv * Ev, wih_b, b_ih, gi, nullptr, 3 * Hv, Ev);
        // gh = h @ w_hh^T + b_hh               (TDM A path)
        wmma_gemm<bf16, false><<<dim3(3 * Hv / BN, Bv / BM), blk, 0, stream>>>(
            h16, Hv, whh_b, b_hh, gh, nullptr, 3 * Hv, Hv);
        // h = (1-z)*n + z*h  (also refresh bf16 mirror)
        gru_gate<<<dim3((Bv * Hv) / 256), blk, 0, stream>>>(gi, gh, hf, h16, Hv);
    }

    // out = h @ agent_W^T + agent_b  (f32 to d_out; TDM A path)
    wmma_gemm<bf16, false><<<dim3(OUTv / BN, Bv / BM), blk, 0, stream>>>(
        h16, Hv, agentW_b, agent_bias, (float*)d_out, nullptr, OUTv, Hv);
}
